// RDGCNEncoder_53953379173285
// MI455X (gfx1250) — matrix operand order repaired and checked by reading, not compiled, not run on hardware
//
#include <hip/hip_runtime.h>
#include <hip/hip_bf16.h>

typedef __attribute__((ext_vector_type(16))) __bf16 v16bf;
typedef __attribute__((ext_vector_type(8)))  float   v8f;

static __device__ __forceinline__ unsigned short f2bf(float f) {
  unsigned u = __float_as_uint(f);
  u += 0x7fffu + ((u >> 16) & 1u);            // round-to-nearest-even
  return (unsigned short)(u >> 16);
}
static __device__ __forceinline__ float bf2f(unsigned short h) {
  return __uint_as_float(((unsigned)h) << 16);
}

// ---------------------------------------------------------------------------
// f32 -> bf16 row-major convert
__global__ __launch_bounds__(256) void cvt_bf16_kernel(const float* __restrict__ in,
                                                       unsigned short* __restrict__ out,
                                                       long long n) {
  long long i = (long long)blockIdx.x * 256 + threadIdx.x;
  if (i < n) out[i] = f2bf(in[i]);
}

// f32 [K][N] -> bf16 transposed [N][K]
__global__ __launch_bounds__(256) void cvt_bf16_t_kernel(const float* __restrict__ in,
                                                         unsigned short* __restrict__ out,
                                                         int K, int N) {
  long long i = (long long)blockIdx.x * 256 + threadIdx.x;
  if (i >= (long long)K * N) return;
  int n = (int)(i / K);
  int k = (int)(i % K);
  out[(size_t)n * K + k] = f2bf(in[(size_t)k * N + n]);
}

__global__ __launch_bounds__(256) void zero_f32_kernel(float* __restrict__ p, long long n) {
  long long i = (long long)blockIdx.x * 256 + threadIdx.x;
  if (i < n) p[i] = 0.0f;
}

// per-node in-degree counts for both directions in one pass
__global__ __launch_bounds__(256) void count_kernel(const int* __restrict__ esrc,
                                                    const int* __restrict__ edst,
                                                    float* __restrict__ cnt_m,
                                                    float* __restrict__ cnt_d, int E) {
  int e = blockIdx.x * 256 + threadIdx.x;
  if (e >= E) return;
  atomicAdd(&cnt_m[esrc[e]], 1.0f);
  atomicAdd(&cnt_d[edst[e]], 1.0f);
}

// gather bf16 source rows per edge, scatter-add f32 onto destinations.
// per-thread = 4 channels; psh = log2(dim/4)
__global__ __launch_bounds__(256) void scatter_add_kernel(const unsigned short* __restrict__ x,
                                                          const int* __restrict__ eidx_src,
                                                          const int* __restrict__ eidx_dst,
                                                          float* __restrict__ agg,
                                                          int E, int dim, int psh) {
  long long idx = (long long)blockIdx.x * 256 + threadIdx.x;
  long long e = idx >> psh;
  if (e >= E) return;
  int c = ((int)idx & ((1 << psh) - 1)) << 2;
  int s = eidx_src[e];
  int d = eidx_dst[e];
  const unsigned short* xp = x + (unsigned)(s * dim + c);   // fits 32-bit
  float* ap = agg + (unsigned)(d * dim + c);
  uint2 w = *(const uint2*)xp;             // 4 bf16
  atomicAdd(ap + 0, bf2f((unsigned short)(w.x & 0xffffu)));
  atomicAdd(ap + 1, bf2f((unsigned short)(w.x >> 16)));
  atomicAdd(ap + 2, bf2f((unsigned short)(w.y & 0xffffu)));
  atomicAdd(ap + 3, bf2f((unsigned short)(w.y >> 16)));
}

// mean = sum / max(cnt,1), cast to bf16; dsh = log2(dim)
__global__ __launch_bounds__(256) void finalize_mean_kernel(const float* __restrict__ agg,
                                                            const float* __restrict__ cnt,
                                                            unsigned short* __restrict__ out,
                                                            long long n, int dsh) {
  long long i = (long long)blockIdx.x * 256 + threadIdx.x;
  if (i >= n) return;
  long long row = i >> dsh;
  float c = cnt[row];
  out[i] = f2bf(agg[i] / (c > 1.0f ? c : 1.0f));
}

// ---------------------------------------------------------------------------
// Fused dual WMMA GEMM: C = A1@B1^T + A2@B2^T + bias (+ addend[idx]) (+relu)
// A*: bf16 row-major [M][K*], B*: bf16 TRANSPOSED [N][K*]
// Block tile 128x128, 8 waves (4x2), each wave 32x64 = 2x4 WMMA 16x16x32 bf16.
// Register-prefetch pipeline via four SCALAR uint4 regs (SROA-safe; no
// promote-alloca-to-LDS). OOB rows/cols are clamped, never masked.
union Frag { v16bf v; unsigned u[8]; };

#define DO_PREFETCH(S)                                                        \
  do {                                                                        \
    const int ss = (S);                                                       \
    const unsigned short* Ap;                                                 \
    const unsigned short* Bp;                                                 \
    int Kp, k0p;                                                              \
    if (ss < steps1) { Ap = A1; Bp = B1; Kp = K1; k0p = ss << 5; }            \
    else             { Ap = A2; Bp = B2; Kp = K2; k0p = (ss - steps1) << 5; } \
    int gr0 = blockM + srow;      gr0 = gr0 < M ? gr0 : M - 1;                \
    int gr1 = blockM + srow + 64; gr1 = gr1 < M ? gr1 : M - 1;                \
    int gn0 = blockN + srow;      gn0 = gn0 < N ? gn0 : N - 1;                \
    int gn1 = blockN + srow + 64; gn1 = gn1 < N ? gn1 : N - 1;                \
    ra0 = *(const uint4*)(Ap + (unsigned)(gr0 * Kp + k0p + skc));             \
    ra1 = *(const uint4*)(Ap + (unsigned)(gr1 * Kp + k0p + skc));             \
    rb0 = *(const uint4*)(Bp + (unsigned)(gn0 * Kp + k0p + skc));             \
    rb1 = *(const uint4*)(Bp + (unsigned)(gn1 * Kp + k0p + skc));             \
  } while (0)

__global__ __launch_bounds__(256)
void wmma_gemm_kernel(const unsigned short* __restrict__ A1, const unsigned short* __restrict__ B1, int K1,
                      const unsigned short* __restrict__ A2, const unsigned short* __restrict__ B2, int K2,
                      const float* __restrict__ bias,
                      const float* __restrict__ addend, const int* __restrict__ addend_idx,
                      float* __restrict__ outF, unsigned short* __restrict__ outBf,
                      int M, int N, int relu) {
  __shared__ unsigned short ldsA[128][40];   // +8 pad: conflict-free fragment reads
  __shared__ unsigned short ldsB[128][40];

  const int tid   = threadIdx.x;
  const int lane  = tid & 31;
  const int wave  = tid >> 5;
  const int waveM = wave >> 1;               // 0..3  (32 rows each)
  const int waveN = wave & 1;                // 0..1  (64 cols each)
  const int half  = lane >> 4;
  const int mr    = lane & 15;
  const int blockM = blockIdx.y * 128;
  const int blockN = blockIdx.x * 128;
  const int srow  = tid >> 2;                // staging row 0..63 (and +64)
  const int skc   = (tid & 3) << 3;          // staging k-offset (8 bf16 chunks)

  // ISA 7.12.2 16-bit fragment K-pair offsets (wave32)
  int kpoA[8], kpoB[8];
#pragma unroll
  for (int v = 0; v < 8; ++v) {
    kpoA[v] = (v < 4) ? (8 * half + 2 * v) : (16 + 8 * half + 2 * (v - 4));
    kpoB[v] = (v < 4) ? (8 * half + 2 * v) : (16 + 8 * half + 2 * (v - 4));
  }

  v8f zero = {0.f, 0.f, 0.f, 0.f, 0.f, 0.f, 0.f, 0.f};
  v8f acc[2][4];
#pragma unroll
  for (int tm = 0; tm < 2; ++tm)
#pragma unroll
    for (int tn = 0; tn < 4; ++tn) acc[tm][tn] = zero;

  const int steps1 = K1 >> 5;
  const int steps2 = A2 ? (K2 >> 5) : 0;
  const int total  = steps1 + steps2;

  uint4 ra0, ra1, rb0, rb1;                  // scalar staging regs (no arrays!)
  DO_PREFETCH(0);

  for (int s = 0; s < total; ++s) {
    *(uint4*)&ldsA[srow][skc]      = ra0;
    *(uint4*)&ldsA[srow + 64][skc] = ra1;
    *(uint4*)&ldsB[srow][skc]      = rb0;
    *(uint4*)&ldsB[srow + 64][skc] = rb1;
    __syncthreads();
    if (s + 1 < total) DO_PREFETCH(s + 1);   // overlap next-step loads with compute

    Frag a[2], b[4];
#pragma unroll
    for (int t = 0; t < 2; ++t) {
      int row = waveM * 32 + t * 16 + mr;
#pragma unroll
      for (int v = 0; v < 8; ++v) a[t].u[v] = *(const unsigned*)&ldsA[row][kpoA[v]];
    }
#pragma unroll
    for (int t = 0; t < 4; ++t) {
      int col = waveN * 64 + t * 16 + mr;
#pragma unroll
      for (int v = 0; v < 8; ++v) b[t].u[v] = *(const unsigned*)&ldsB[col][kpoB[v]];
    }
#pragma unroll
    for (int tm = 0; tm < 2; ++tm)
#pragma unroll
      for (int tn = 0; tn < 4; ++tn)
        acc[tm][tn] = __builtin_amdgcn_wmma_f32_16x16x32_bf16(
            false, a[tm].v, false, b[tn].v, (short)0, acc[tm][tn], false, false);
    __syncthreads();
  }

  // epilogue: bias + gathered addend + relu; write f32 and/or bf16
#pragma unroll
  for (int tm = 0; tm < 2; ++tm) {
#pragma unroll
    for (int tn = 0; tn < 4; ++tn) {
      int gn = blockN + waveN * 64 + tn * 16 + mr;
#pragma unroll
      for (int r = 0; r < 8; ++r) {
        int gm = blockM + waveM * 32 + tm * 16 + half * 8 + r;
        if (gm < M && gn < N) {
          float v = acc[tm][tn][r];
          if (bias) v += bias[gn];
          if (addend) {
            int arow = addend_idx ? addend_idx[gm] : gm;
            v += addend[(unsigned)(arow * N + gn)];
          }
          if (relu && v < 0.f) v = 0.f;
          unsigned co = (unsigned)(gm * N + gn);
          if (outF)  outF[co] = v;
          if (outBf) outBf[co] = f2bf(v);
        }
      }
    }
  }
}

// ---------------------------------------------------------------------------
extern "C" void kernel_launch(void* const* d_in, const int* in_sizes, int n_in,
                              void* d_out, int out_size, void* d_ws, size_t ws_size,
                              hipStream_t stream) {
  const float* mirna_feat = (const float*)d_in[0];
  const float* dis_feat   = (const float*)d_in[1];
  const int*   mirna_id   = (const int*)d_in[2];
  const int*   dis_id     = (const int*)d_in[3];
  const int*   edge_src   = (const int*)d_in[4];
  const int*   edge_dst   = (const int*)d_in[5];
  const float* Wm = (const float*)d_in[6];  const float* bm = (const float*)d_in[7];
  const float* Wd = (const float*)d_in[8];  const float* bd = (const float*)d_in[9];
  const float* mirna_emb = (const float*)d_in[10];
  const float* dis_emb   = (const float*)d_in[11];
  const float* W1_md_l = (const float*)d_in[12]; const float* b1_md = (const float*)d_in[13];
  const float* W1_md_r = (const float*)d_in[14];
  const float* W1_dm_l = (const float*)d_in[15]; const float* b1_dm = (const float*)d_in[16];
  const float* W1_dm_r = (const float*)d_in[17];
  const float* W2_md_l = (const float*)d_in[18]; const float* b2_md = (const float*)d_in[19];
  const float* W2_md_r = (const float*)d_in[20];
  const float* W2_dm_l = (const float*)d_in[21]; const float* b2_dm = (const float*)d_in[22];
  const float* W2_dm_r = (const float*)d_in[23];

  const int NM  = in_sizes[2];
  const int ND  = in_sizes[3];
  const int E   = in_sizes[4];
  const int FIN = in_sizes[0] / NM;
  const int H   = in_sizes[7];    // 512
  const int H1  = in_sizes[13];   // 512
  const int O   = in_sizes[19];   // 256

  float* out_m = (float*)d_out;                     // [NM, O]
  float* out_d = (float*)d_out + (size_t)NM * O;    // [ND, O]

  // ---- workspace carve (256B aligned) ----
  char* ws = (char*)d_ws;
  size_t off = 0;
  auto carve = [&](size_t bytes) -> void* {
    void* p = ws + off;
    off = (off + bytes + 255) & ~(size_t)255;
    return p;
  };
  unsigned short* wm_t    = (unsigned short*)carve((size_t)H  * FIN * 2);
  unsigned short* wd_t    = (unsigned short*)carve((size_t)H  * FIN * 2);
  unsigned short* w1mdl_t = (unsigned short*)carve((size_t)H1 * H * 2);
  unsigned short* w1mdr_t = (unsigned short*)carve((size_t)H1 * H * 2);
  unsigned short* w1dml_t = (unsigned short*)carve((size_t)H1 * H * 2);
  unsigned short* w1dmr_t = (unsigned short*)carve((size_t)H1 * H * 2);
  unsigned short* w2mdl_t = (unsigned short*)carve((size_t)O  * H1 * 2);
  unsigned short* w2mdr_t = (unsigned short*)carve((size_t)O  * H1 * 2);
  unsigned short* w2dml_t = (unsigned short*)carve((size_t)O  * H1 * 2);
  unsigned short* w2dmr_t = (unsigned short*)carve((size_t)O  * H1 * 2);
  unsigned short* featm_bf = (unsigned short*)carve((size_t)NM * FIN * 2);
  unsigned short* featd_bf = (unsigned short*)carve((size_t)ND * FIN * 2);
  unsigned short* xm_bf    = (unsigned short*)carve((size_t)NM * H * 2);
  unsigned short* xd_bf    = (unsigned short*)carve((size_t)ND * H * 2);
  unsigned short* hm_bf    = (unsigned short*)carve((size_t)NM * H1 * 2);
  unsigned short* hd_bf    = (unsigned short*)carve((size_t)ND * H1 * 2);
  unsigned short* aggm_bf  = (unsigned short*)carve((size_t)NM * H * 2);
  unsigned short* aggd_bf  = (unsigned short*)carve((size_t)ND * H * 2);
  float* aggF  = (float*)carve((size_t)NM * H * 4);   // reused sequentially
  float* cnt_m = (float*)carve((size_t)NM * 4);
  float* cnt_d = (float*)carve((size_t)ND * 4);

  auto g1 = [](long long n) { return dim3((unsigned)((n + 255) / 256)); };

  // ---- weight conversion (bf16, transposed [N][K]) ----
  cvt_bf16_t_kernel<<<g1((long long)FIN * H), 256, 0, stream>>>(Wm, wm_t, FIN, H);
  cvt_bf16_t_kernel<<<g1((long long)FIN * H), 256, 0, stream>>>(Wd, wd_t, FIN, H);
  cvt_bf16_t_kernel<<<g1((long long)H * H1), 256, 0, stream>>>(W1_md_l, w1mdl_t, H, H1);
  cvt_bf16_t_kernel<<<g1((long long)H * H1), 256, 0, stream>>>(W1_md_r, w1mdr_t, H, H1);
  cvt_bf16_t_kernel<<<g1((long long)H * H1), 256, 0, stream>>>(W1_dm_l, w1dml_t, H, H1);
  cvt_bf16_t_kernel<<<g1((long long)H * H1), 256, 0, stream>>>(W1_dm_r, w1dmr_t, H, H1);
  cvt_bf16_t_kernel<<<g1((long long)H1 * O), 256, 0, stream>>>(W2_md_l, w2mdl_t, H1, O);
  cvt_bf16_t_kernel<<<g1((long long)H1 * O), 256, 0, stream>>>(W2_md_r, w2mdr_t, H1, O);
  cvt_bf16_t_kernel<<<g1((long long)H1 * O), 256, 0, stream>>>(W2_dm_l, w2dml_t, H1, O);
  cvt_bf16_t_kernel<<<g1((long long)H1 * O), 256, 0, stream>>>(W2_dm_r, w2dmr_t, H1, O);
  cvt_bf16_kernel<<<g1((long long)NM * FIN), 256, 0, stream>>>(mirna_feat, featm_bf, (long long)NM * FIN);
  cvt_bf16_kernel<<<g1((long long)ND * FIN), 256, 0, stream>>>(dis_feat, featd_bf, (long long)ND * FIN);

  auto gemm = [&](const unsigned short* A1, const unsigned short* B1, int K1,
                  const unsigned short* A2, const unsigned short* B2, int K2,
                  const float* bias, const float* addend, const int* aidx,
                  float* oF, unsigned short* oB, int M, int N, int relu) {
    dim3 grid((N + 127) / 128, (M + 127) / 128);
    wmma_gemm_kernel<<<grid, 256, 0, stream>>>(A1, B1, K1, A2, B2, K2, bias,
                                               addend, aidx, oF, oB, M, N, relu);
  };

  // ---- encoders: x = feat @ W + b + emb[id]  (bf16 out) ----
  gemm(featm_bf, wm_t, FIN, nullptr, nullptr, 0, bm, mirna_emb, mirna_id,
       nullptr, xm_bf, NM, H, 0);
  gemm(featd_bf, wd_t, FIN, nullptr, nullptr, 0, bd, dis_emb, dis_id,
       nullptr, xd_bf, ND, H, 0);

  // ---- degree counts (shared by both layers) ----
  zero_f32_kernel<<<g1(NM), 256, 0, stream>>>(cnt_m, NM);
  zero_f32_kernel<<<g1(ND), 256, 0, stream>>>(cnt_d, ND);
  count_kernel<<<g1(E), 256, 0, stream>>>(edge_src, edge_dst, cnt_m, cnt_d, E);

  const int pshH  = __builtin_ctz(H >> 2);
  const int pshH1 = __builtin_ctz(H1 >> 2);
  const int dshH  = __builtin_ctz(H);
  const int dshH1 = __builtin_ctz(H1);
  const long long edH  = (long long)E << pshH;
  const long long edH1 = (long long)E << pshH1;

  // ---- layer 1, disease side: h_d = relu(mean(x_m) @ W1l + b + x_d @ W1r) ----
  zero_f32_kernel<<<g1((long long)ND * H), 256, 0, stream>>>(aggF, (long long)ND * H);
  scatter_add_kernel<<<g1(edH), 256, 0, stream>>>(xm_bf, edge_src, edge_dst, aggF, E, H, pshH);
  finalize_mean_kernel<<<g1((long long)ND * H), 256, 0, stream>>>(aggF, cnt_d, aggd_bf, (long long)ND * H, dshH);
  gemm(aggd_bf, w1mdl_t, H, xd_bf, w1mdr_t, H, b1_md, nullptr, nullptr,
       nullptr, hd_bf, ND, H1, 1);

  // ---- layer 1, mirna side ----
  zero_f32_kernel<<<g1((long long)NM * H), 256, 0, stream>>>(aggF, (long long)NM * H);
  scatter_add_kernel<<<g1(edH), 256, 0, stream>>>(xd_bf, edge_dst, edge_src, aggF, E, H, pshH);
  finalize_mean_kernel<<<g1((long long)NM * H), 256, 0, stream>>>(aggF, cnt_m, aggm_bf, (long long)NM * H, dshH);
  gemm(aggm_bf, w1dml_t, H, xm_bf, w1dmr_t, H, b1_dm, nullptr, nullptr,
       nullptr, hm_bf, NM, H1, 1);

  // ---- layer 2, disease side: o_d -> d_out tail ----
  zero_f32_kernel<<<g1((long long)ND * H1), 256, 0, stream>>>(aggF, (long long)ND * H1);
  scatter_add_kernel<<<g1(edH1), 256, 0, stream>>>(hm_bf, edge_src, edge_dst, aggF, E, H1, pshH1);
  finalize_mean_kernel<<<g1((long long)ND * H1), 256, 0, stream>>>(aggF, cnt_d, aggd_bf, (long long)ND * H1, dshH1);
  gemm(aggd_bf, w2mdl_t, H1, hd_bf, w2mdr_t, H1, b2_md, nullptr, nullptr,
       out_d, nullptr, ND, O, 0);

  // ---- layer 2, mirna side: o_m -> d_out head ----
  zero_f32_kernel<<<g1((long long)NM * H1), 256, 0, stream>>>(aggF, (long long)NM * H1);
  scatter_add_kernel<<<g1(edH1), 256, 0, stream>>>(hd_bf, edge_dst, edge_src, aggF, E, H1, pshH1);
  finalize_mean_kernel<<<g1((long long)NM * H1), 256, 0, stream>>>(aggF, cnt_m, aggm_bf, (long long)NM * H1, dshH1);
  gemm(aggm_bf, w2dml_t, H1, hm_bf, w2dmr_t, H1, b2_dm, nullptr, nullptr,
       out_m, nullptr, NM, O, 0);
}
